// O3TensorProductWeighted_50440095924877
// MI455X (gfx1250) — compile-verified, matrix operand order checked by reading, not run on hardware
//
#include <hip/hip_runtime.h>
#include <hip/hip_bf16.h>

// ---------------------------------------------------------------------------
// MI455X fused implementation, round 3.
//
// All math through v_wmma_f32_16x16x32_f16, per-element factors folded into
// the A operand. Round-3: (a) each wave owns TWO 16-element tiles (M=32) so
// every B block feeds 4 WMMAs (halves L1/L2 demand per FLOP); (b) the three
// Cartesian Wc GEMMs share one B pass (12 WMMAs per B block). 4-wave WGs,
// all waves stream the same B addresses => WGP$ reuse; L2 traffic ~153MB.
// ---------------------------------------------------------------------------

typedef _Float16 v16h __attribute__((ext_vector_type(16)));
typedef float    v8f  __attribute__((ext_vector_type(8)));

#define N1_ 16384
#define N2_ 8192
#define N3_ 4096
#define WNUMEL_ 36864
#define NROW_ 4096
#define OUTC_ 320

// all outer scales cancel: PW0/SQRT_K == 1, PW1*INV_SQRT3/SQRT_K == 1
#define INV_SQRT3_ 0.57735026918962576f

// sec0 = Wa: U=128, NC=128, KT=260   sec1 = Wd: U=64, NC=128, KT=130
// sec2 = Wb: U=128, NC=64,  KT=260   sec3 = Wc: U=64, NC=64,  KT=130
#define SZA_ ((size_t)8320*128)
#define SZD_ ((size_t)4160*128)
#define SZB_ ((size_t)8320*64)
#define SZC_ ((size_t)4160*64)
#define BTOT_ (SZA_+SZD_+SZB_+SZC_)      /* 2,396,160 halves = 4.79 MB */

// workspace layout (in halves)
#define OFF_H_    (BTOT_)                               /* h:   [4096][64]       */
#define OFF_S1T_  (OFF_H_   + (size_t)4096*64)          /* s1T: [256][128][16]   */
#define OFF_S1R_  (OFF_S1T_ + (size_t)4096*128)         /* s1R: [4096][128]      */
#define OFF_D12T_ (OFF_S1R_ + (size_t)4096*128)         /* d12T:[256][64][16]    */
#define OFF_D12R_ (OFF_D12T_+ (size_t)4096*64)          /* d12R:[4096][64]       */
#define OFF_V1T_  (OFF_D12R_+ (size_t)4096*64)          /* v1T: [3][256][64][16] */
#define OFF_V1R_  (OFF_V1T_ + (size_t)3*4096*64)        /* v1R: [3][4096][64]    */

static __device__ inline v8f wmma16(v16h a, v16h b, v8f c) {
    return __builtin_amdgcn_wmma_f32_16x16x32_f16(
        false, a, false, b, (short)0, c, false, false);
}

// ---------------------------------------------------------------------------
// Repack W2 (64 x 36864 f32) + b2 into f16 B-operand images.
// Block = 512 halves; lane L owns halves [L*16..L*16+15] = column nt*16+(L&15),
// K rows kt*32 + (L>>4)*16 + j (K-contiguous per lane).
// ---------------------------------------------------------------------------
__global__ __launch_bounds__(256) void repack_w2(
    const float* __restrict__ W2, const float* __restrict__ b2,
    _Float16* __restrict__ B)
{
    size_t idx = (size_t)blockIdx.x * 256 + threadIdx.x;
    if (idx >= BTOT_) return;
    size_t off = idx;
    int sec;
    if (off < SZA_)                sec = 0;
    else if ((off -= SZA_) < SZD_) sec = 1;
    else if ((off -= SZD_) < SZB_) sec = 2;
    else { off -= SZB_;            sec = 3; }

    const int U  = (sec == 0 || sec == 2) ? 128 : 64;
    const int KT = 2 * U + U / 32;

    int within = (int)(off & 511);
    size_t blk = off >> 9;
    int kt   = (int)(blk % (size_t)KT);
    int nt   = (int)(blk / (size_t)KT);
    int lane = within >> 4;
    int j    = within & 15;

    int r = kt * 32 + (lane >> 4) * 16 + j;   // K row
    int c = nt * 16 + (lane & 15);            // N col
    int Kreg = U * 64;
    int u = (r < Kreg) ? (r >> 6) : (r - Kreg);

    int colbase;
    if (sec == 0)      colbase = u * 128;                    // Wa
    else if (sec == 1) colbase = (N1_+N2_+N3_) + u * 128;    // Wd
    else if (sec == 2) colbase = N1_ + u * 64;               // Wb
    else               colbase = (N1_+N2_) + u * 64;         // Wc

    float val = (r < Kreg) ? W2[(size_t)(r & 63) * WNUMEL_ + colbase + c]
                           : b2[colbase + c];
    B[idx] = (_Float16)val;
}

// ---------------------------------------------------------------------------
// Prep: per-element MLP h (f16) + contraction factors in two layouts:
//   *T : [tile][u][16 m]  -> contiguous 32B per-u broadcast reads
//   *R : [e][u]           -> contiguous rows for bias-row A operands
// ---------------------------------------------------------------------------
__global__ __launch_bounds__(256) void o3tp_prep(
    const float* __restrict__ in1, const float* __restrict__ in2,
    const float* __restrict__ wt,  const float* __restrict__ W0,
    const float* __restrict__ b0,  const float* __restrict__ W1,
    const float* __restrict__ b1,  _Float16* __restrict__ ws)
{
    __shared__ float sh0[16 * 64];
    const int tid  = threadIdx.x;
    const int tile = blockIdx.x;
    const int e0   = tile * 16;

    _Float16* hG   = ws + OFF_H_;
    _Float16* s1T  = ws + OFF_S1T_;
    _Float16* s1R  = ws + OFF_S1R_;
    _Float16* d12T = ws + OFF_D12T_;
    _Float16* d12R = ws + OFF_D12R_;
    _Float16* v1T  = ws + OFF_V1T_;
    _Float16* v1R  = ws + OFF_V1R_;

    for (int idx = tid; idx < 16 * 128; idx += 256) {
        int m = idx >> 7, u = idx & 127;
        _Float16 v = (_Float16)in1[(size_t)(e0 + m) * OUTC_ + u];
        s1T[(size_t)tile * 2048 + u * 16 + m] = v;
        s1R[(size_t)(e0 + m) * 128 + u]       = v;
    }
    for (int idx = tid; idx < 16 * 64; idx += 256) {
        int m = idx >> 6, u = idx & 63;
        const float* p  = &in1[(size_t)(e0 + m) * OUTC_ + 128 + u * 3];
        const float* i2 = &in2[(size_t)(e0 + m) * 4];
        float a0 = p[0], a1 = p[1], a2 = p[2];
        float d  = a0 * i2[1] + a1 * i2[2] + a2 * i2[3];
        v1T[((size_t)0 * 256 + tile) * 1024 + u * 16 + m] = (_Float16)a0;
        v1T[((size_t)1 * 256 + tile) * 1024 + u * 16 + m] = (_Float16)a1;
        v1T[((size_t)2 * 256 + tile) * 1024 + u * 16 + m] = (_Float16)a2;
        v1R[((size_t)0 * 4096 + e0 + m) * 64 + u] = (_Float16)a0;
        v1R[((size_t)1 * 4096 + e0 + m) * 64 + u] = (_Float16)a1;
        v1R[((size_t)2 * 4096 + e0 + m) * 64 + u] = (_Float16)a2;
        d12T[(size_t)tile * 1024 + u * 16 + m] = (_Float16)d;
        d12R[(size_t)(e0 + m) * 64 + u]        = (_Float16)d;
    }
    for (int idx = tid; idx < 16 * 64; idx += 256) {
        int m = idx >> 6, j = idx & 63;
        float acc = b0[j];
        const float* wrow = &wt[(size_t)(e0 + m) * 16];
#pragma unroll
        for (int i = 0; i < 16; ++i) acc += wrow[i] * W0[i * 64 + j];
        sh0[m * 64 + j] = fmaxf(acc, 0.f);
    }
    __syncthreads();
    for (int idx = tid; idx < 16 * 64; idx += 256) {
        int m = idx >> 6, k = idx & 63;
        float acc = b1[k];
        for (int j = 0; j < 64; ++j) acc += sh0[m * 64 + j] * W1[j * 64 + k];
        hG[(size_t)(e0 + m) * 64 + k] = (_Float16)fmaxf(acc, 0.f);
    }
}

// ---------------------------------------------------------------------------
// Two element-tiles (M=32) against one B stream: 4 WMMA per B block.
// ---------------------------------------------------------------------------
template <int U>
static __device__ inline void gemm2(
    v16h ha0a, v16h ha1a, v16h ha0b, v16h ha1b,
    const _Float16* __restrict__ fTa, const _Float16* __restrict__ fRa,
    const _Float16* __restrict__ fTb, const _Float16* __restrict__ fRb,
    const v16h* __restrict__ Bl, int lh, v8f& accA, v8f& accB)
{
#pragma unroll 4
    for (int u = 0; u < U; ++u) {
        if ((u & 7) == 0)
            __builtin_prefetch((const void*)(Bl + (size_t)(2 * u + 16) * 32), 0, 0);
        _Float16 fa = fTa[u * 16];
        _Float16 fb = fTb[u * 16];
        v16h b0 = Bl[(size_t)(2 * u) * 32];
        v16h b1 = Bl[(size_t)(2 * u) * 32 + 32];
        accA = wmma16(ha0a * fa, b0, accA);
        accA = wmma16(ha1a * fa, b1, accA);
        accB = wmma16(ha0b * fb, b0, accB);
        accB = wmma16(ha1b * fb, b1, accB);
    }
#pragma unroll
    for (int t = 0; t < U / 32; ++t) {
        v16h bB = Bl[(size_t)(2 * U + t) * 32];
        accA = wmma16(*(const v16h*)&fRa[t * 32 + lh * 16], bB, accA);
        accB = wmma16(*(const v16h*)&fRb[t * 32 + lh * 16], bB, accB);
    }
}

// ---------------------------------------------------------------------------
// Two element-tiles x 3 Cartesian components against one B stream:
// 12 WMMA per B block (Wc section, U=64).
// ---------------------------------------------------------------------------
static __device__ inline void gemm2x3(
    v16h ha0a, v16h ha1a, v16h ha0b, v16h ha1b,
    const _Float16* __restrict__ fTa0, const _Float16* __restrict__ fRa0,
    const _Float16* __restrict__ fTa1, const _Float16* __restrict__ fRa1,
    const _Float16* __restrict__ fTa2, const _Float16* __restrict__ fRa2,
    const _Float16* __restrict__ fTb0, const _Float16* __restrict__ fRb0,
    const _Float16* __restrict__ fTb1, const _Float16* __restrict__ fRb1,
    const _Float16* __restrict__ fTb2, const _Float16* __restrict__ fRb2,
    const v16h* __restrict__ Bl, int lh,
    v8f& a0c, v8f& a1c, v8f& a2c, v8f& b0c, v8f& b1c, v8f& b2c)
{
#pragma unroll 2
    for (int u = 0; u < 64; ++u) {
        if ((u & 7) == 0)
            __builtin_prefetch((const void*)(Bl + (size_t)(2 * u + 16) * 32), 0, 0);
        v16h b0 = Bl[(size_t)(2 * u) * 32];
        v16h b1 = Bl[(size_t)(2 * u) * 32 + 32];
        _Float16 f;
        f = fTa0[u * 16]; a0c = wmma16(ha0a * f, b0, a0c); a0c = wmma16(ha1a * f, b1, a0c);
        f = fTa1[u * 16]; a1c = wmma16(ha0a * f, b0, a1c); a1c = wmma16(ha1a * f, b1, a1c);
        f = fTa2[u * 16]; a2c = wmma16(ha0a * f, b0, a2c); a2c = wmma16(ha1a * f, b1, a2c);
        f = fTb0[u * 16]; b0c = wmma16(ha0b * f, b0, b0c); b0c = wmma16(ha1b * f, b1, b0c);
        f = fTb1[u * 16]; b1c = wmma16(ha0b * f, b0, b1c); b1c = wmma16(ha1b * f, b1, b1c);
        f = fTb2[u * 16]; b2c = wmma16(ha0b * f, b0, b2c); b2c = wmma16(ha1b * f, b1, b2c);
    }
#pragma unroll
    for (int t = 0; t < 2; ++t) {
        v16h bB = Bl[(size_t)(128 + t) * 32];
        a0c = wmma16(*(const v16h*)&fRa0[t * 32 + lh * 16], bB, a0c);
        a1c = wmma16(*(const v16h*)&fRa1[t * 32 + lh * 16], bB, a1c);
        a2c = wmma16(*(const v16h*)&fRa2[t * 32 + lh * 16], bB, a2c);
        b0c = wmma16(*(const v16h*)&fRb0[t * 32 + lh * 16], bB, b0c);
        b1c = wmma16(*(const v16h*)&fRb1[t * 32 + lh * 16], bB, b1c);
        b2c = wmma16(*(const v16h*)&fRb2[t * 32 + lh * 16], bB, b2c);
    }
}

// ---------------------------------------------------------------------------
// Main: grid = 32 supertiles x 12 jobs, 128 threads (4 waves x 2 tiles).
// jobs 0..7: out0 tile nt=job (Wa + Wd); jobs 8..11: out1 tile (Wb + fused Wc).
// ---------------------------------------------------------------------------
__global__ __launch_bounds__(128) void o3tp_main(
    const float* __restrict__ in2, const float* __restrict__ biasv,
    const _Float16* __restrict__ ws, float* __restrict__ out)
{
    const int job = blockIdx.x % 12;
    const int st  = blockIdx.x / 12;
    const int wv = threadIdx.x >> 5, lane = threadIdx.x & 31;
    const int ll = lane & 15, lh = lane >> 4;
    const int e0a = st * 128 + wv * 32;
    const int e0b = e0a + 16;
    const int ta  = e0a >> 4, tb = ta + 1;

    const _Float16* hG = ws + OFF_H_;
    v16h ha0a = *(const v16h*)&hG[(size_t)(e0a + ll) * 64 + lh * 16];
    v16h ha1a = *(const v16h*)&hG[(size_t)(e0a + ll) * 64 + 32 + lh * 16];
    v16h ha0b = *(const v16h*)&hG[(size_t)(e0b + ll) * 64 + lh * 16];
    v16h ha1b = *(const v16h*)&hG[(size_t)(e0b + ll) * 64 + 32 + lh * 16];

    const _Float16* s1Ta  = ws + OFF_S1T_  + (size_t)ta * 2048 + ll;
    const _Float16* s1Tb  = ws + OFF_S1T_  + (size_t)tb * 2048 + ll;
    const _Float16* s1Ra  = ws + OFF_S1R_  + (size_t)(e0a + ll) * 128;
    const _Float16* s1Rb  = ws + OFF_S1R_  + (size_t)(e0b + ll) * 128;

    const v16h* Bv = (const v16h*)ws;
    const v16h* BA = Bv;                              // KT=260, NC=128
    const v16h* BD = Bv + SZA_ / 16;                  // KT=130, NC=128
    const v16h* BB = Bv + (SZA_ + SZD_) / 16;         // KT=260, NC=64
    const v16h* BC = Bv + (SZA_ + SZD_ + SZB_) / 16;  // KT=130, NC=64

    if (job < 8) {
        const int nt = job;
        const _Float16* dTa = ws + OFF_D12T_ + (size_t)ta * 1024 + ll;
        const _Float16* dTb = ws + OFF_D12T_ + (size_t)tb * 1024 + ll;
        const _Float16* dRa = ws + OFF_D12R_ + (size_t)(e0a + ll) * 64;
        const _Float16* dRb = ws + OFF_D12R_ + (size_t)(e0b + ll) * 64;
        v8f caA = {}, caB = {}, cdA = {}, cdB = {};
        gemm2<128>(ha0a, ha1a, ha0b, ha1b, s1Ta, s1Ra, s1Tb, s1Rb,
                   BA + (size_t)nt * 260 * 32 + lane, lh, caA, caB);
        gemm2<64>(ha0a, ha1a, ha0b, ha1b, dTa, dRa, dTb, dRb,
                  BD + (size_t)nt * 130 * 32 + lane, lh, cdA, cdB);
        const int w = nt * 16 + ll;
        const float bw = biasv[w];
#pragma unroll
        for (int vr = 0; vr < 8; ++vr) {
            const int mr = vr + 8 * lh;
            int e = e0a + mr;
            float val = in2[(size_t)e * 4] * caA[vr] + INV_SQRT3_ * cdA[vr] + bw;
            __builtin_nontemporal_store(val, &out[(size_t)e * OUTC_ + w]);
            e = e0b + mr;
            val = in2[(size_t)e * 4] * caB[vr] + INV_SQRT3_ * cdB[vr] + bw;
            __builtin_nontemporal_store(val, &out[(size_t)e * OUTC_ + w]);
        }
    } else {
        const int nt = job - 8;
        v8f ctA = {}, ctB = {};
        gemm2<128>(ha0a, ha1a, ha0b, ha1b, s1Ta, s1Ra, s1Tb, s1Rb,
                   BB + (size_t)nt * 260 * 32 + lane, lh, ctA, ctB);
        v8f cA0 = {}, cA1 = {}, cA2 = {}, cB0 = {}, cB1 = {}, cB2 = {};
        gemm2x3(ha0a, ha1a, ha0b, ha1b,
                ws + OFF_V1T_ + ((size_t)0 * 256 + ta) * 1024 + ll,
                ws + OFF_V1R_ + ((size_t)0 * 4096 + e0a + ll) * 64,
                ws + OFF_V1T_ + ((size_t)1 * 256 + ta) * 1024 + ll,
                ws + OFF_V1R_ + ((size_t)1 * 4096 + e0a + ll) * 64,
                ws + OFF_V1T_ + ((size_t)2 * 256 + ta) * 1024 + ll,
                ws + OFF_V1R_ + ((size_t)2 * 4096 + e0a + ll) * 64,
                ws + OFF_V1T_ + ((size_t)0 * 256 + tb) * 1024 + ll,
                ws + OFF_V1R_ + ((size_t)0 * 4096 + e0b + ll) * 64,
                ws + OFF_V1T_ + ((size_t)1 * 256 + tb) * 1024 + ll,
                ws + OFF_V1R_ + ((size_t)1 * 4096 + e0b + ll) * 64,
                ws + OFF_V1T_ + ((size_t)2 * 256 + tb) * 1024 + ll,
                ws + OFF_V1R_ + ((size_t)2 * 4096 + e0b + ll) * 64,
                BC + (size_t)nt * 130 * 32 + lane, lh,
                cA0, cA1, cA2, cB0, cB1, cB2);
        const int w = nt * 16 + ll;
#pragma unroll
        for (int vr = 0; vr < 8; ++vr) {
            const int mr = vr + 8 * lh;
            int e = e0a + mr;
            const float* i2 = &in2[(size_t)e * 4];
            float* op = &out[(size_t)e * OUTC_ + 128 + w * 3];
            __builtin_nontemporal_store(ctA[vr] * i2[1] + i2[0] * cA0[vr], op + 0);
            __builtin_nontemporal_store(ctA[vr] * i2[2] + i2[0] * cA1[vr], op + 1);
            __builtin_nontemporal_store(ctA[vr] * i2[3] + i2[0] * cA2[vr], op + 2);
            e = e0b + mr;
            i2 = &in2[(size_t)e * 4];
            op = &out[(size_t)e * OUTC_ + 128 + w * 3];
            __builtin_nontemporal_store(ctB[vr] * i2[1] + i2[0] * cB0[vr], op + 0);
            __builtin_nontemporal_store(ctB[vr] * i2[2] + i2[0] * cB1[vr], op + 1);
            __builtin_nontemporal_store(ctB[vr] * i2[3] + i2[0] * cB2[vr], op + 2);
        }
    }
}

// ---------------------------------------------------------------------------
extern "C" void kernel_launch(void* const* d_in, const int* in_sizes, int n_in,
                              void* d_out, int out_size, void* d_ws, size_t ws_size,
                              hipStream_t stream)
{
    const float* in1   = (const float*)d_in[0];   // (4096, 320)
    const float* in2   = (const float*)d_in[1];   // (4096, 4)
    const float* wt    = (const float*)d_in[2];   // (4096, 16)
    const float* W0    = (const float*)d_in[3];   // (16, 64)
    const float* b0    = (const float*)d_in[4];   // (64)
    const float* W1    = (const float*)d_in[5];   // (64, 64)
    const float* b1    = (const float*)d_in[6];   // (64)
    const float* W2    = (const float*)d_in[7];   // (64, 36864)
    const float* b2    = (const float*)d_in[8];   // (36864)
    const float* biasv = (const float*)d_in[9];   // (128)
    float*       out   = (float*)d_out;
    _Float16*    ws    = (_Float16*)d_ws;         // ~11.6 MB used

    (void)in_sizes; (void)n_in; (void)out_size; (void)ws_size;

    repack_w2<<<(unsigned)(BTOT_ / 256), 256, 0, stream>>>(W2, b2, ws);
    o3tp_prep<<<NROW_ / 16, 256, 0, stream>>>(in1, in2, wt, W0, b0, W1, b1, ws);
    o3tp_main<<<32 * 12, 128, 0, stream>>>(in2, biasv, ws, out);
}
